// AttentionAggregator_48601849921795
// MI455X (gfx1250) — matrix operand order repaired and checked by reading, not compiled, not run on hardware
//
#include <hip/hip_runtime.h>
#include <hip/hip_bf16.h>
#include <math.h>

// ---------------------------------------------------------------------------
// AttentionAggregator for MI455X (gfx1250, wave32, WMMA bf16 16x16x32)
// ---------------------------------------------------------------------------

typedef __attribute__((ext_vector_type(16))) __bf16 v16bf;
typedef __attribute__((ext_vector_type(8)))  float  v8f;

union Frag {
    uint4 q[2];   // 8 dwords = 16 bf16 per lane
    v16bf v;
};

#define D_DIM    256
#define H_DIM    128
#define TILE_M   128
#define A_STRIDE 264                       // ushorts per A row: 256 + 8 pad (528 B, 4-dword bank rotation)
#define B_STRIDE 136                       // ushorts per W1 row: 128 + 8 pad (272 B)
#define A_BYTES  (TILE_M * A_STRIDE * 2)   // 67584
#define B_BYTES  (D_DIM  * B_STRIDE * 2)   // 69632
#define SMEM_BYTES (A_BYTES + B_BYTES)     // 137216 (< 320 KB; 2 WGs/WGP)

// CDNA5 hardware tanh (TRANS op, co-executes with WMMA)
__device__ inline float tanh_hw(float x) {
    float y;
    asm("v_tanh_f32 %0, %1" : "=v"(y) : "v"(x));
    return y;
}
// CDNA5 packed f32 -> bf16 convert (single VALU op)
__device__ inline unsigned int pack_bf16x2(float lo, float hi) {
    unsigned int r;
    asm("v_cvt_pk_bf16_f32 %0, %1, %2" : "=v"(r) : "v"(lo), "v"(hi));
    return r;
}
// order-preserving float -> uint encoding for atomic segment max
__device__ inline unsigned int enc_f32(float f) {
    unsigned int u = __float_as_uint(f);
    return (u & 0x80000000u) ? ~u : (u | 0x80000000u);
}
__device__ inline float dec_f32(unsigned int e) {
    unsigned int u = (e & 0x80000000u) ? (e ^ 0x80000000u) : ~e;
    return __uint_as_float(u);
}

// ---------------------------------------------------------------------------
// Kernel 0: zero the per-segment max (encoded; 0 == "empty") and denominators
// ---------------------------------------------------------------------------
__global__ void aa_init_kernel(unsigned int* __restrict__ segmax_enc,
                               float* __restrict__ denom, int B) {
    int i = blockIdx.x * blockDim.x + threadIdx.x;
    if (i < B) { segmax_enc[i] = 0u; denom[i] = 0.0f; }
}

// ---------------------------------------------------------------------------
// Kernel 1: scores = tanh(X@W1 + b1)@W2 + b2  via bf16 WMMA, + atomic seg-max
//   block: 256 threads (8 waves), 128 node rows; wave w owns M-tile rows
//   [16w, 16w+16), sweeps all 8 H-tiles with K=256 in 8 WMMA steps each.
// ---------------------------------------------------------------------------
__global__ void __launch_bounds__(256)
aa_score_kernel(const float* __restrict__ X,
                const unsigned char* __restrict__ mask,
                const int* __restrict__ batch,
                const float* __restrict__ W1,
                const float* __restrict__ b1,
                const float* __restrict__ W2,
                const float* __restrict__ b2,
                float* __restrict__ scores,
                unsigned int* __restrict__ segmax_enc,
                int N) {
    extern __shared__ __align__(32) char smem[];
    unsigned short* Abuf = (unsigned short*)smem;             // [128][264] bf16
    unsigned short* Bbuf = (unsigned short*)(smem + A_BYTES); // [256][136] bf16

    const int t  = threadIdx.x;
    const int m0 = blockIdx.x * TILE_M;

    // ---- stage X tile (f32 -> bf16), float4 loads: thread t -> col4 t&63 ----
    {
        const int c4 = t & 63;
        const int rb = t >> 6;
        #pragma unroll 4
        for (int i = 0; i < 32; ++i) {
            int r  = i * 4 + rb;
            int gm = m0 + r; if (gm >= N) gm = N - 1;          // clamp tail
            float4 v = *(const float4*)(X + (size_t)gm * D_DIM + c4 * 4);
            unsigned int* dst = (unsigned int*)(Abuf + r * A_STRIDE + c4 * 4);
            dst[0] = pack_bf16x2(v.x, v.y);
            dst[1] = pack_bf16x2(v.z, v.w);
        }
    }
    // ---- stage W1 (f32 -> bf16), row-major [K=256][H=128] ----
    {
        const int n4 = t & 31;
        const int kb = t >> 5;
        #pragma unroll 4
        for (int i = 0; i < 32; ++i) {
            int k = i * 8 + kb;
            float4 v = *(const float4*)(W1 + (size_t)k * H_DIM + n4 * 4);
            unsigned int* dst = (unsigned int*)(Bbuf + k * B_STRIDE + n4 * 4);
            dst[0] = pack_bf16x2(v.x, v.y);
            dst[1] = pack_bf16x2(v.z, v.w);
        }
    }
    __syncthreads();

    const int wave = t >> 5;
    const int lane = t & 31;
    const int hf   = lane >> 4;    // half-wave index
    const int lrow = lane & 15;

    // ---- A fragments for this wave's 16 rows, all 8 K-steps ----
    // ISA A-layout (16-bit 16x32): lane holds row M=lane%16; VGPR0..3 = K pairs
    // starting at k0+8*hf, VGPR4..7 = K pairs starting at k0+16+8*hf.
    Frag afr[8];
    {
        const unsigned short* arow = Abuf + (wave * 16 + lrow) * A_STRIDE;
        #pragma unroll
        for (int kk = 0; kk < 8; ++kk) {
            const int k0 = kk * 32;
            afr[kk].q[0] = *(const uint4*)(arow + k0 + 8 * hf);
            afr[kk].q[1] = *(const uint4*)(arow + k0 + 16 + 8 * hf);
        }
    }

    float sacc[8];
    #pragma unroll
    for (int j = 0; j < 8; ++j) sacc[j] = 0.0f;

    #pragma unroll
    for (int nt = 0; nt < 8; ++nt) {
        v8f c = {};
        #pragma unroll
        for (int kk = 0; kk < 8; ++kk) {
            // ISA B-layout (16-bit 32x16): lane holds column N=n0+lane%16; elements
            // 0..7 = K k0+8*hf.., elements 8..15 = K k0+16+8*hf.. (matches A).
            Frag bfr;
            const unsigned short* bcol = Bbuf + (kk * 32) * B_STRIDE + nt * 16 + lrow;
            unsigned short* bs = (unsigned short*)bfr.q;
            #pragma unroll
            for (int s = 0; s < 8; ++s) {
                bs[s]     = bcol[(8 * hf + s) * B_STRIDE];
                bs[8 + s] = bcol[(16 + 8 * hf + s) * B_STRIDE];
            }
            c = __builtin_amdgcn_wmma_f32_16x16x32_bf16(
                    false, afr[kk].v, false, bfr.v, (short)0, c, false, false);
        }
        // C layout: VGPR j -> (M = j + 8*hf, N = nt*16 + lrow)
        const int h   = nt * 16 + lrow;
        const float b1v = b1[h];
        const float w2v = W2[h];
        #pragma unroll
        for (int j = 0; j < 8; ++j)
            sacc[j] += tanh_hw(c[j] + b1v) * w2v;
    }

    // reduce over the 16 lanes of each half-wave (N dimension of C)
    #pragma unroll
    for (int m = 1; m < 16; m <<= 1) {
        #pragma unroll
        for (int j = 0; j < 8; ++j)
            sacc[j] += __shfl_xor(sacc[j], m, 32);
    }

    if (lrow == 0) {   // lanes 0 and 16: rows j and j+8 of this wave's M-tile
        const float b2v = b2[0];
        #pragma unroll
        for (int j = 0; j < 8; ++j) {
            int gm = m0 + wave * 16 + hf * 8 + j;
            if (gm < N) {
                float s = sacc[j] + b2v;
                if (!mask[gm]) s = -INFINITY;
                scores[gm] = s;
                atomicMax(&segmax_enc[batch[gm]], enc_f32(s));
            }
        }
    }
}

// ---------------------------------------------------------------------------
// Kernel 2: e = mask ? exp(s - segmax) : 0 (in place); denom += e
// ---------------------------------------------------------------------------
__global__ void aa_exp_kernel(float* __restrict__ scores_e,
                              const int* __restrict__ batch,
                              const unsigned char* __restrict__ mask,
                              const unsigned int* __restrict__ segmax_enc,
                              float* __restrict__ denom, int N) {
    int n = blockIdx.x * blockDim.x + threadIdx.x;
    if (n >= N) return;
    int b = batch[n];
    unsigned int me = segmax_enc[b];
    float m = (me == 0u) ? 0.0f : dec_f32(me);
    if (!isfinite(m)) m = 0.0f;                 // matches reference seg_max fixup
    float e = mask[n] ? __expf(scores_e[n] - m) : 0.0f;
    scores_e[n] = e;
    atomicAdd(&denom[b], e);
}

// ---------------------------------------------------------------------------
// Kernel 3: one block per segment; binary-search [lo,hi) in sorted batch,
// thread d accumulates column d; out[b,d] = (sum_n x[n,d]*e[n]) / denom[b]
// ---------------------------------------------------------------------------
__device__ inline int lower_bound_i32(const int* __restrict__ a, int n, int key) {
    int lo = 0, hi = n;
    while (lo < hi) { int mid = (lo + hi) >> 1; if (a[mid] < key) lo = mid + 1; else hi = mid; }
    return lo;
}

__global__ void __launch_bounds__(256)
aa_agg_kernel(const float* __restrict__ X,
              const float* __restrict__ e,
              const int* __restrict__ batch,
              const float* __restrict__ denom,
              float* __restrict__ out, int N) {
    __shared__ int seg[2];
    const int b = blockIdx.x;
    if (threadIdx.x == 0) {
        seg[0] = lower_bound_i32(batch, N, b);
        seg[1] = lower_bound_i32(batch, N, b + 1);
    }
    __syncthreads();
    const int lo = seg[0], hi = seg[1];
    const float dn  = denom[b];
    const float inv = dn > 0.0f ? 1.0f / dn : 0.0f;
    const int d = threadIdx.x;

    float acc = 0.0f;
    int n = lo;
    for (; n + 3 < hi; n += 4) {
        float w0 = e[n], w1 = e[n + 1], w2 = e[n + 2], w3 = e[n + 3];
        acc += X[(size_t)(n    ) * D_DIM + d] * w0;
        acc += X[(size_t)(n + 1) * D_DIM + d] * w1;
        acc += X[(size_t)(n + 2) * D_DIM + d] * w2;
        acc += X[(size_t)(n + 3) * D_DIM + d] * w3;
    }
    for (; n < hi; ++n) acc += X[(size_t)n * D_DIM + d] * e[n];

    out[(size_t)b * D_DIM + d] = acc * inv;   // empty segment -> 0, as reference
}

// ---------------------------------------------------------------------------
extern "C" void kernel_launch(void* const* d_in, const int* in_sizes, int n_in,
                              void* d_out, int out_size, void* d_ws, size_t ws_size,
                              hipStream_t stream) {
    const float*         X    = (const float*)d_in[0];
    const int*           batch= (const int*)d_in[1];
    const unsigned char* mask = (const unsigned char*)d_in[2];
    const float*         W1   = (const float*)d_in[3];
    const float*         b1   = (const float*)d_in[4];
    const float*         W2   = (const float*)d_in[5];
    const float*         b2   = (const float*)d_in[6];
    float*               out  = (float*)d_out;

    const int N = in_sizes[1];          // batch has N elements
    const int B = out_size / D_DIM;     // 1024

    // workspace layout: [segmax_enc: B u32][denom: B f32][scores/e: N f32]
    char* ws = (char*)d_ws;
    unsigned int* segmax_enc = (unsigned int*)ws;
    float*        denom      = (float*)(ws + 4096);
    float*        scores_e   = (float*)(ws + 8192);

    aa_init_kernel<<<(B + 255) / 256, 256, 0, stream>>>(segmax_enc, denom, B);

    const int nblk1 = (N + TILE_M - 1) / TILE_M;
    aa_score_kernel<<<nblk1, 256, SMEM_BYTES, stream>>>(
        X, mask, batch, W1, b1, W2, b2, scores_e, segmax_enc, N);

    aa_exp_kernel<<<(N + 255) / 256, 256, 0, stream>>>(
        scores_e, batch, mask, segmax_enc, denom, N);

    aa_agg_kernel<<<B, 256, 0, stream>>>(X, scores_e, batch, denom, out, N);
}
